// EncoderLayer_22067541966976
// MI455X (gfx1250) — matrix operand order repaired and checked by reading, not compiled, not run on hardware
//
#include <hip/hip_runtime.h>
#include <math.h>

// ---------------------------------------------------------------------------
// CDNA5 (gfx1250) Evoformer block. wave32, WMMA f32<=f16 16x16x32.
// All GEMM tiles are exact (M%16==0, N%16==0, K%32==0 guaranteed by padding).
// AV path: A tile double-buffered in LDS via GLOBAL_LOAD_ASYNC_TO_LDS_B128
// (ASYNCcnt pipeline); B tile cooperatively converted to f16 in LDS.
// ---------------------------------------------------------------------------

typedef __attribute__((ext_vector_type(16))) _Float16 v16h;
typedef __attribute__((ext_vector_type(8)))  float    v8f;
typedef __attribute__((ext_vector_type(4)))  float    v4f;

#define WPB 4  // waves per block (128 threads); 4 M-tiles share one B tile

__device__ __forceinline__ float gelu_exact(float x) {
  return 0.5f * x * (1.0f + erff(x * 0.70710678118654752440f));
}

template <int N>
__device__ __forceinline__ void wait_asynccnt() {
#if __has_builtin(__builtin_amdgcn_s_wait_asynccnt)
  __builtin_amdgcn_s_wait_asynccnt(N);
#else
  asm volatile("s_wait_asynccnt %0" :: "i"(N) : "memory");
#endif
}

// Async DMA: 16 bytes global -> LDS, tracked by ASYNCcnt (ISA 15.18.3 op 98).
// ldsOff is the 32-bit LDS byte offset (low 32 bits of the flat shared addr).
__device__ __forceinline__ void async_g2l_b128(unsigned ldsOff, const float* gp) {
  asm volatile("global_load_async_to_lds_b128 %0, %1, off"
               :: "v"(ldsOff), "v"(gp) : "memory");
}

// Universal strided WMMA GEMM:
//   acc[m,n] = sum_k A[m*rsA + k*csA + z*bsA] * B[k*rsB + n*csB + z*bsB]
//   t = act(acc*cscale + bias[n]) * rowscale[m] * (*gscale) * emul[m,n,z]
//   C[m*rsC + n*csC + z*bsC] = t + resid[m*rsR + n*csR + z*bsR]
// act: 0 none, 1 exact gelu, 2 sigmoid. Any pointer may be null (skipped).
// REQUIRES: M%16==0, N%16==0, K%32==0. AV => csA==1, BV => csB==1 (16B-aligned).
template <bool AV, bool BV>
__global__ __launch_bounds__(32 * WPB) void wmma_gemm(
    const float* __restrict__ A, int rsA, int csA, long bsA,
    const float* __restrict__ B, int rsB, int csB, long bsB,
    float* __restrict__ C, int rsC, int csC, long bsC,
    const float* __restrict__ bias,
    const float* __restrict__ resid, int rsR, int csR, long bsR,
    const float* __restrict__ rowscale,
    const float* __restrict__ gscale,
    const float* __restrict__ emul, long bsE,
    float cscale, int act,
    int M, int N, int K)
{
  __shared__ alignas(32) _Float16 Bs[512];             // 32x16 f16 B tile
  __shared__ alignas(16) float As[AV ? 2 * WPB * 16 * 32 : 4]; // dbl-buf A f32

  const int tid  = threadIdx.x;
  const int lane = tid & 31;
  const int wave = tid >> 5;
  const int tilesM = (M + 15) >> 4;
  int tm = blockIdx.y * WPB + wave;
  if (tm >= tilesM) tm = tilesM - 1;         // uniform clamp (keeps barriers safe)
  const int tn = blockIdx.x;
  const int bz = blockIdx.z;
  const int m0 = tm << 4;
  const int n0 = tn << 4;
  const int half = lane >> 4;                // 0: lanes 0-15, 1: lanes 16-31
  const int l16  = lane & 15;

  const float* __restrict__ Ab = A + (long)bz * bsA;   // uniform (SGPR) bases
  const float* __restrict__ Bb = B + (long)bz * bsB;

  // A 16x32 f16 operand layout (ISA 7.12.2): elem e -> k = (e/8)*16 + half*8 + e%8
  const int am = m0 + l16;                   // A row owned by this lane
  int aOff[16];
  if constexpr (!AV) {
    const int aRowOff = am * rsA;
#pragma unroll
    for (int e = 0; e < 16; ++e)
      aOff[e] = aRowOff + ((((e >> 3) << 4) + (half << 3)) + (e & 7)) * csA;
  }

  // AV: issue one stage of async A copies (4 x b128 per lane) into buffer bsel.
  // Wave-private region: As[(bsel*WPB + wave)*512 ..], rows r=0..15, k=0..31.
  auto issueA = [&](int k0, int bsel) {
#pragma unroll
    for (int i = 0; i < 4; ++i) {
      const int chunk = (i << 5) + lane;     // 0..127 (16B chunks of 16x32 tile)
      const int r  = chunk >> 3;             // 0..15
      const int c4 = (chunk & 7) << 2;       // 0,4,..,28
      const float* gp = Ab + (m0 + r) * rsA + k0 + c4;
      const unsigned ldsOff =
          (unsigned)(size_t)&As[((bsel * WPB + wave) << 9) + (r << 5) + c4];
      async_g2l_b128(ldsOff, gp);
    }
  };

  if constexpr (AV) issueA(0, 0);

  v8f acc = {0.f, 0.f, 0.f, 0.f, 0.f, 0.f, 0.f, 0.f};

  int s = 0;
  for (int k0 = 0; k0 < K; k0 += 32, ++s) {
    // ---- cooperative B tile -> LDS (f32->f16 before store), shared by 4 waves.
    // B 32x16 f16 operand layout: lane = n + ((k/8)%2)*16, elem e = (k/16)*8 + k%8.
    if constexpr (BV) {
      const int kk  = tid >> 2;              // 0..31
      const int nn0 = (tid & 3) << 2;        // 0,4,8,12
      const v4f b4 = *(const v4f*)(Bb + (k0 + kk) * rsB + n0 + nn0);
#pragma unroll
      for (int u = 0; u < 4; ++u) {
        const int nn = nn0 + u;
        Bs[((nn + (((kk >> 3) & 1) << 4)) << 4) + ((kk >> 4) << 3) + (kk & 7)] = (_Float16)b4[u];
      }
    } else {
#pragma unroll
      for (int i = 0; i < 4; ++i) {
        const int idx = tid + (i << 7);      // 0..511
        const int kk  = idx >> 4;
        const int nn  = idx & 15;
        const float fv = Bb[(k0 + kk) * rsB + (n0 + nn) * csB];
        Bs[((nn + (((kk >> 3) & 1) << 4)) << 4) + ((kk >> 4) << 3) + (kk & 7)] = (_Float16)fv;
      }
    }

    // ---- AV: software pipeline the A DMA one stage ahead.
    if constexpr (AV) {
      if (k0 + 32 < K) { issueA(k0 + 32, (s + 1) & 1); wait_asynccnt<4>(); }
      else             { wait_asynccnt<0>(); }
    }
    __syncthreads();
    const v16h bv = *((const v16h*)&Bs[lane << 4]);   // 2x ds_load_b128

    // ---- A fragment ----
    v16h av;
    if constexpr (AV) {
      const float* lA = &As[(((s & 1) * WPB + wave) << 9) + (l16 << 5) + (half << 3)];
      const v4f a0 = *(const v4f*)(lA);
      const v4f a1 = *(const v4f*)(lA + 4);
      const v4f a2 = *(const v4f*)(lA + 16);
      const v4f a3 = *(const v4f*)(lA + 20);
#pragma unroll
      for (int u = 0; u < 4; ++u) {
        av[u]      = (_Float16)a0[u];
        av[u + 4]  = (_Float16)a1[u];
        av[u + 8]  = (_Float16)a2[u];
        av[u + 12] = (_Float16)a3[u];
      }
    } else {
      const int kbase = k0 * csA;
#pragma unroll
      for (int e = 0; e < 16; ++e)
        av[e] = (_Float16)Ab[aOff[e] + kbase];
      if (k0 + 32 < K) __builtin_prefetch(&Ab[aOff[0] + kbase + 32 * csA], 0, 1);
    }

    acc = __builtin_amdgcn_wmma_f32_16x16x32_f16(
        /*neg_a=*/false, av, /*neg_b=*/false, bv,
        /*c_mod=*/(short)0, acc, /*reuse_a=*/false, /*reuse_b=*/false);
    __syncthreads();                          // before next B fill / A overwrite
  }

  // ---- epilogue (once per tile). C/D layout: VGPR r -> M = half*8 + r.
  const int n = n0 + l16;
  const float bv0 = bias ? bias[n] : 0.f;
#pragma unroll
  for (int r = 0; r < 8; ++r) {
    const int m = m0 + (half << 3) + r;
    float t = acc[r] * cscale + bv0;
    if (act == 1)      t = gelu_exact(t);
    else if (act == 2) t = 1.f / (1.f + __expf(-t));
    if (rowscale) t *= rowscale[m];
    if (gscale)   t *= gscale[0];
    const long ci = (long)m * rsC + (long)n * csC;
    if (emul)  t *= emul[ci + (long)bz * bsE];
    if (resid) t += resid[(long)m * rsR + (long)n * csR + (long)bz * bsR];
    C[ci + (long)bz * bsC] = t;
  }
}

// LayerNorm over last dim L; one block per row.
template <int BS>
__global__ void ln_kernel(const float* __restrict__ x,
                          const float* __restrict__ g,
                          const float* __restrict__ b,
                          float* __restrict__ y, int L)
{
  __shared__ float red[BS];
  const long row = blockIdx.x;
  const float* xr = x + row * (long)L;
  float* yr = y + row * (long)L;

  float s = 0.f;
  for (int i = threadIdx.x; i < L; i += BS) s += xr[i];
  red[threadIdx.x] = s; __syncthreads();
  for (int st = BS >> 1; st > 0; st >>= 1) {
    if ((int)threadIdx.x < st) red[threadIdx.x] += red[threadIdx.x + st];
    __syncthreads();
  }
  const float mu = red[0] / (float)L;
  __syncthreads();

  float v = 0.f;
  for (int i = threadIdx.x; i < L; i += BS) { const float d = xr[i] - mu; v += d * d; }
  red[threadIdx.x] = v; __syncthreads();
  for (int st = BS >> 1; st > 0; st >>= 1) {
    if ((int)threadIdx.x < st) red[threadIdx.x] += red[threadIdx.x + st];
    __syncthreads();
  }
  const float inv = rsqrtf(red[0] / (float)L + 1e-5f);
  for (int i = threadIdx.x; i < L; i += BS)
    yr[i] = (xr[i] - mu) * inv * g[i] + b[i];
}

// Numerically stable softmax over last dim L, in place; one block per row.
template <int BS>
__global__ void softmax_kernel(float* __restrict__ x, int L)
{
  __shared__ float red[BS];
  const long row = blockIdx.x;
  float* xr = x + row * (long)L;

  float mx = -3.4e38f;
  for (int i = threadIdx.x; i < L; i += BS) mx = fmaxf(mx, xr[i]);
  red[threadIdx.x] = mx; __syncthreads();
  for (int st = BS >> 1; st > 0; st >>= 1) {
    if ((int)threadIdx.x < st) red[threadIdx.x] = fmaxf(red[threadIdx.x], red[threadIdx.x + st]);
    __syncthreads();
  }
  mx = red[0];
  __syncthreads();

  float s = 0.f;
  for (int i = threadIdx.x; i < L; i += BS) { const float e = __expf(xr[i] - mx); xr[i] = e; s += e; }
  red[threadIdx.x] = s; __syncthreads();
  for (int st = BS >> 1; st > 0; st >>= 1) {
    if ((int)threadIdx.x < st) red[threadIdx.x] += red[threadIdx.x + st];
    __syncthreads();
  }
  const float inv = 1.f / red[0];
  for (int i = threadIdx.x; i < L; i += BS) xr[i] *= inv;
}

__global__ void ew_mul(const float* __restrict__ a, const float* __restrict__ b,
                       float* __restrict__ o, long n)
{
  const long i = blockIdx.x * (long)blockDim.x + threadIdx.x;
  if (i < n) o[i] = a[i] * b[i];
}

// ab2 = P * sigmoid_already_applied(G) * pair_mask[row] * n^-0.5  (64 chans/row)
__global__ void ew_combine_ab2(const float* __restrict__ P, const float* __restrict__ G,
                               const float* __restrict__ pm, float* __restrict__ out,
                               long n, float invs)
{
  const long i = blockIdx.x * (long)blockDim.x + threadIdx.x;
  if (i < n) { const long m = i >> 6; out[i] = P[i] * G[i] * pm[m] * invs; }
}

// W2[d, c*64+p] = wout[c*32+d, p]   (32 x 2048 from 1024 x 64)
__global__ void repack_wout(const float* __restrict__ wout, float* __restrict__ W2)
{
  const int i = blockIdx.x * blockDim.x + threadIdx.x;
  if (i < 32 * 2048) {
    const int d = i >> 11, r = i & 2047, c = r >> 6, p = r & 63;
    W2[i] = wout[(c * 32 + d) * 64 + p];
  }
}

// Pad wpb [64,8]->[64,16] and bpb [8]->[16] with zeros so the pair-bias GEMM
// runs on exact 16-wide tiles with no bounds checks.
__global__ void pad_wpb(const float* __restrict__ wpb, const float* __restrict__ bpb,
                        float* __restrict__ wpb_p, float* __restrict__ bpb_p)
{
  const int i = blockIdx.x * blockDim.x + threadIdx.x;
  if (i < 1024) { const int r = i >> 4, c = i & 15; wpb_p[i] = (c < 8) ? wpb[r * 8 + c] : 0.f; }
  if (i < 16)   bpb_p[i] = (i < 8) ? bpb[i] : 0.f;
}

// ---------------------------------------------------------------------------

static inline void launch_gemm(hipStream_t s,
    const float* A, int rsA, int csA, long bsA,
    const float* B, int rsB, int csB, long bsB,
    float* C, int rsC, int csC, long bsC,
    const float* bias,
    const float* resid, int rsR, int csR, long bsR,
    const float* rowscale, const float* gscale,
    const float* emul, long bsE,
    float cscale, int act, int M, int N, int K, int batch)
{
  const int tilesM = (M + 15) / 16;
  dim3 grid((N + 15) / 16, (tilesM + WPB - 1) / WPB, batch);
  dim3 blk(32 * WPB);
  if (csA == 1 && csB == 1)
    wmma_gemm<true, true><<<grid, blk, 0, s>>>(A, rsA, csA, bsA, B, rsB, csB, bsB,
        C, rsC, csC, bsC, bias, resid, rsR, csR, bsR, rowscale, gscale, emul, bsE,
        cscale, act, M, N, K);
  else if (csA == 1)
    wmma_gemm<true, false><<<grid, blk, 0, s>>>(A, rsA, csA, bsA, B, rsB, csB, bsB,
        C, rsC, csC, bsC, bias, resid, rsR, csR, bsR, rowscale, gscale, emul, bsE,
        cscale, act, M, N, K);
  else if (csB == 1)
    wmma_gemm<false, true><<<grid, blk, 0, s>>>(A, rsA, csA, bsA, B, rsB, csB, bsB,
        C, rsC, csC, bsC, bias, resid, rsR, csR, bsR, rowscale, gscale, emul, bsE,
        cscale, act, M, N, K);
  else
    wmma_gemm<false, false><<<grid, blk, 0, s>>>(A, rsA, csA, bsA, B, rsB, csB, bsB,
        C, rsC, csC, bsC, bias, resid, rsR, csR, bsR, rowscale, gscale, emul, bsE,
        cscale, act, M, N, K);
}

extern "C" void kernel_launch(void* const* d_in, const int* in_sizes, int n_in,
                              void* d_out, int out_size, void* d_ws, size_t ws_size,
                              hipStream_t stream)
{
  (void)in_sizes; (void)n_in; (void)out_size; (void)ws_size;

  // ---- inputs (setup_inputs order) ----
  const float* node_in   = (const float*)d_in[0];   // [384,768]
  const float* pair_in   = (const float*)d_in[1];   // [384,384,64]
  const float* pair_mask = (const float*)d_in[2];   // [384,384]
  const float* attn_mask = (const float*)d_in[3];   // [1,1,1,384]
  const float* op_mask   = (const float*)d_in[4];   // [384,1]
  const float* op_norm   = (const float*)d_in[5];   // scalar
  int pi = 6;
  const float* ln_attn_g = (const float*)d_in[pi++];
  const float* ln_attn_b = (const float*)d_in[pi++];
  const float* wq        = (const float*)d_in[pi++];
  const float* wk        = (const float*)d_in[pi++];
  const float* wv        = (const float*)d_in[pi++];
  const float* wg        = (const float*)d_in[pi++];
  const float* bg        = (const float*)d_in[pi++];
  const float* wo        = (const float*)d_in[pi++];
  const float* bo        = (const float*)d_in[pi++];
  const float* wpb       = (const float*)d_in[pi++];
  const float* bpb       = (const float*)d_in[pi++];
  const float* ln_fin_g  = (const float*)d_in[pi++];
  const float* ln_fin_b  = (const float*)d_in[pi++];
  const float* ffn_w1    = (const float*)d_in[pi++];
  const float* ffn_b1    = (const float*)d_in[pi++];
  const float* ffn_w2    = (const float*)d_in[pi++];
  const float* ffn_b2    = (const float*)d_in[pi++];
  const float* ln_opm_g  = (const float*)d_in[pi++];
  const float* ln_opm_b  = (const float*)d_in[pi++];
  const float* opm_win   = (const float*)d_in[pi++];
  const float* opm_bin   = (const float*)d_in[pi++];
  const float* opm_wout  = (const float*)d_in[pi++];
  const float* opm_bout  = (const float*)d_in[pi++];
  const float* ln_tm_g   = (const float*)d_in[pi++];
  const float* ln_tm_b   = (const float*)d_in[pi++];
  const float* tm_wab_p  = (const float*)d_in[pi++];
  const float* tm_bab_p  = (const float*)d_in[pi++];
  const float* tm_wab_g  = (const float*)d_in[pi++];
  const float* tm_bab_g  = (const float*)d_in[pi++];
  const float* tm_wg     = (const float*)d_in[pi++];
  const float* tm_bg     = (const float*)d_in[pi++];
  const float* tm_ln_g   = (const float*)d_in[pi++];
  const float* tm_ln_b   = (const float*)d_in[pi++];
  const float* tm_wz     = (const float*)d_in[pi++];
  const float* tm_bz     = (const float*)d_in[pi++];
  const float* ln_pf_g   = (const float*)d_in[pi++];
  const float* ln_pf_b   = (const float*)d_in[pi++];
  const float* pf_w1     = (const float*)d_in[pi++];
  const float* pf_b1     = (const float*)d_in[pi++];
  const float* pf_w2     = (const float*)d_in[pi++];
  const float* pf_b2     = (const float*)d_in[pi++];

  const long ND = 384L * 768;        // 294912
  const long PP = 384L * 384 * 64;   // 9437184
  const long XX = 384L * 384 * 32;   // 4718592

  // ---- workspace carve-up ----
  float* W = (float*)d_ws;
  size_t off = 0;
  auto alloc = [&](size_t n) { float* p = W + off; off += n; return p; };
  float* qn    = alloc(ND);
  float* q     = alloc(ND);
  float* k     = alloc(ND);
  float* v     = alloc(ND);
  float* gt    = alloc(ND);
  float* pb    = alloc(384L * 384 * 16);  // pair bias [i,j,h] padded to 16 heads
  float* att   = alloc(384L * 384 * 8);   // logits -> attn [h,i,j]
  float* o     = alloc(ND);
  float* og    = alloc(ND);
  float* node1 = alloc(ND);
  float* h1    = alloc(384L * 3072);
  float* ln2   = alloc(ND);
  float* ab    = alloc(384L * 64);
  float* W2    = alloc(32L * 2048);
  float* T     = alloc(384L * 2048);
  float* wpb_p = alloc(64L * 16);
  float* bpb_p = alloc(16);
  float* pair1 = alloc(PP);
  float* zt    = alloc(PP);               // later reused as pair2
  float* g2    = alloc(PP);
  float* Pb    = alloc(PP);               // later reused: ab2, then hp
  float* Gb    = alloc(PP);               // later reused: pffn hidden
  float* xb    = alloc(XX);
  float* xln   = alloc(XX);

  float* node_out = (float*)d_out;                 // [384,768]
  float* pair_out = (float*)d_out + ND;            // [384,384,64]

  const float qscale  = 0.102062072615966f;        // 96^-0.5
  const float trimask = 0.051031036307983f;        // 384^-0.5

  // ================= node track =================
  ln_kernel<256><<<384, 256, 0, stream>>>(node_in, ln_attn_g, ln_attn_b, qn, 768);
  launch_gemm(stream, qn,768,1,0, wq,768,1,0, q,768,1,0, nullptr, nullptr,0,0,0,
              nullptr,nullptr,nullptr,0, 1.f,0, 384,768,768, 1);
  launch_gemm(stream, qn,768,1,0, wk,768,1,0, k,768,1,0, nullptr, nullptr,0,0,0,
              nullptr,nullptr,nullptr,0, 1.f,0, 384,768,768, 1);
  launch_gemm(stream, qn,768,1,0, wv,768,1,0, v,768,1,0, nullptr, nullptr,0,0,0,
              nullptr,nullptr,nullptr,0, 1.f,0, 384,768,768, 1);
  launch_gemm(stream, qn,768,1,0, wg,768,1,0, gt,768,1,0, bg, nullptr,0,0,0,
              nullptr,nullptr,nullptr,0, 1.f,2, 384,768,768, 1);
  // pair bias (padded): pb[i,j,h16] = pair @ wpb_p + bpb_p  (M=384^2, N=16, K=64)
  pad_wpb<<<8, 128, 0, stream>>>(wpb, bpb, wpb_p, bpb_p);
  launch_gemm(stream, pair_in,64,1,0, wpb_p,16,1,0, pb,16,1,0, bpb_p, nullptr,0,0,0,
              nullptr,nullptr,nullptr,0, 1.f,0, 147456,16,64, 1);
  // logits[h,i,j] = qscale*(q.k) + attn_mask[j] + pb[i,j,h]   (batched over h)
  launch_gemm(stream, q,768,1,96, k,1,768,96, att,384,1,147456, attn_mask,
              pb,6144,16,1, nullptr,nullptr,nullptr,0, qscale,0, 384,384,96, 8);
  softmax_kernel<128><<<3072, 128, 0, stream>>>(att, 384);
  // o[i, h*96+d] = attn @ v   (batched over h)
  launch_gemm(stream, att,384,1,147456, v,768,1,96, o,768,1,96, nullptr,
              nullptr,0,0,0, nullptr,nullptr,nullptr,0, 1.f,0, 384,96,384, 8);
  ew_mul<<<(int)((ND + 255) / 256), 256, 0, stream>>>(gt, o, og, ND);
  // node1 = node + (gate*o) @ wo + bo
  launch_gemm(stream, og,768,1,0, wo,768,1,0, node1,768,1,0, bo,
              node_in,768,1,0, nullptr,nullptr,nullptr,0, 1.f,0, 384,768,768, 1);
  // FFN
  ln_kernel<256><<<384, 256, 0, stream>>>(node1, ln_fin_g, ln_fin_b, ln2, 768);
  launch_gemm(stream, ln2,768,1,0, ffn_w1,3072,1,0, h1,3072,1,0, ffn_b1,
              nullptr,0,0,0, nullptr,nullptr,nullptr,0, 1.f,1, 384,3072,768, 1);
  launch_gemm(stream, h1,3072,1,0, ffn_w2,768,1,0, node_out,768,1,0, ffn_b2,
              node1,768,1,0, nullptr,nullptr,nullptr,0, 1.f,0, 384,768,3072, 1);

  // ================= outer-product-mean (factored) =================
  ln_kernel<256><<<384, 256, 0, stream>>>(node_out, ln_opm_g, ln_opm_b, ln2, 768);
  launch_gemm(stream, ln2,768,1,0, opm_win,64,1,0, ab,64,1,0, opm_bin,
              nullptr,0,0,0, op_mask,nullptr,nullptr,0, 1.f,0, 384,64,768, 1);
  repack_wout<<<(32 * 2048 + 255) / 256, 256, 0, stream>>>(opm_wout, W2);
  // T[j, c*64+p] = b_[j,:] @ W2
  launch_gemm(stream, ab + 32,64,1,0, W2,2048,1,0, T,2048,1,0, nullptr,
              nullptr,0,0,0, nullptr,nullptr,nullptr,0, 1.f,0, 384,2048,32, 1);
  // pair1[i,j,p] = pair + ((a_[i,:] @ T[j]) + bout) * op_norm   (batched over j)
  launch_gemm(stream, ab,64,1,0, T,64,1,2048, pair1,24576,1,64, opm_bout,
              pair_in,24576,1,64, nullptr, op_norm, nullptr,0, 1.f,0, 384,64,32, 384);

  // ================= triangle multiplicative update =================
  ln_kernel<64><<<147456, 64, 0, stream>>>(pair1, ln_tm_g, ln_tm_b, zt, 64);
  launch_gemm(stream, zt,64,1,0, tm_wg,64,1,0, g2,64,1,0, tm_bg,
              nullptr,0,0,0, nullptr,nullptr,nullptr,0, 1.f,2, 147456,64,64, 1);
  launch_gemm(stream, zt,64,1,0, tm_wab_p,64,1,0, Pb,64,1,0, tm_bab_p,
              nullptr,0,0,0, nullptr,nullptr,nullptr,0, 1.f,0, 147456,64,64, 1);
  launch_gemm(stream, zt,64,1,0, tm_wab_g,64,1,0, Gb,64,1,0, tm_bab_g,
              nullptr,0,0,0, nullptr,nullptr,nullptr,0, 1.f,2, 147456,64,64, 1);
  ew_combine_ab2<<<(int)((PP + 255) / 256), 256, 0, stream>>>(Pb, Gb, pair_mask, Pb, PP, trimask);
  // outgoing: x[i,j,c] = sum_k a2[i,k,c] * b2[j,k,c]    (batched over c=0..31)
  launch_gemm(stream, Pb,24576,64,1, Pb + 32,64,24576,1, xb,12288,32,1, nullptr,
              nullptr,0,0,0, nullptr,nullptr,nullptr,0, 1.f,0, 384,384,384, 32);
  // incoming: x += sum_k a2[k,i,c] * b2[k,j,c]
  launch_gemm(stream, Pb,64,24576,1, Pb + 32,24576,64,1, xb,12288,32,1, nullptr,
              xb,12288,32,1, nullptr,nullptr,nullptr,0, 1.f,0, 384,384,384, 32);
  ln_kernel<32><<<147456, 32, 0, stream>>>(xb, tm_ln_g, tm_ln_b, xln, 32);
  // pair2 (in zt) = pair1 + g2 * (xln @ tm_wz + tm_bz)
  launch_gemm(stream, xln,32,1,0, tm_wz,64,1,0, zt,64,1,0, tm_bz,
              pair1,64,1,0, nullptr,nullptr, g2,0, 1.f,0, 147456,64,32, 1);

  // ================= pair FFN =================
  ln_kernel<64><<<147456, 64, 0, stream>>>(zt, ln_pf_g, ln_pf_b, Pb, 64);
  launch_gemm(stream, Pb,64,1,0, pf_w1,64,1,0, Gb,64,1,0, pf_b1,
              nullptr,0,0,0, nullptr,nullptr,nullptr,0, 1.f,1, 147456,64,64, 1);
  launch_gemm(stream, Gb,64,1,0, pf_w2,64,1,0, pair_out,64,1,0, pf_b2,
              zt,64,1,0, nullptr,nullptr,nullptr,0, 1.f,0, 147456,64,64, 1);
}
